// DeepSeekMoEGate_22797686407759
// MI455X (gfx1250) — matrix operand order repaired and checked by reading, not compiled, not run on hardware
//
#include <hip/hip_runtime.h>
#include <hip/hip_bf16.h>
#include <stdint.h>

#define NUM_EXPERTS 256
#define TOP_K       8
#define N_GROUP     8
#define TOPK_GROUP  4
#define HIDDEN      7168
#define TOKENS      16384
#define ROUTED_SCALING 2.5f
#define TOK_PER_BLOCK 64

typedef __bf16    v16bf __attribute__((ext_vector_type(16)));
typedef float     v8f   __attribute__((ext_vector_type(8)));
typedef float     f32x4 __attribute__((ext_vector_type(4)));
typedef uint32_t  u32x4 __attribute__((ext_vector_type(4)));

// Pack two f32 -> bf16x2 in one VALU op where possible.
#if __has_builtin(__builtin_amdgcn_cvt_pk_bf16_f32)
typedef __bf16 v2bf __attribute__((ext_vector_type(2)));
__device__ __forceinline__ uint32_t pack2_bf16(float lo, float hi) {
  union { v2bf v; uint32_t u; } cv;
  cv.v = __builtin_amdgcn_cvt_pk_bf16_f32(lo, hi);   // v_cvt_pk_bf16_f32 (RNE)
  return cv.u;
}
#else
__device__ __forceinline__ uint32_t pack2_bf16(float lo, float hi) {
  // single v_perm_b32: {hi[31:16], lo[31:16]} (truncating bf16)
  return __builtin_amdgcn_perm(__float_as_uint(hi), __float_as_uint(lo),
                               0x07060302u);
}
#endif

// ---------------- pre-pass: weight f32 -> bf16 (3.7 MB, lives in L2) -------
__global__ void weight_to_bf16(const float* __restrict__ w,
                               uint32_t* __restrict__ o, int npairs) {
  int i = blockIdx.x * blockDim.x + threadIdx.x;
  if (i < npairs) o[i] = pack2_bf16(w[2 * i], w[2 * i + 1]);
}

// ---------------- fused GEMM (bf16 WMMA, f32 acc) + sigmoid + routing ------
__global__ __launch_bounds__(512) void moe_gate(
    const float* __restrict__ hidden,       // [TOKENS, HIDDEN] f32
    const uint16_t* __restrict__ wbf,       // [NUM_EXPERTS, HIDDEN] bf16
    const float* __restrict__ bias,         // [NUM_EXPERTS] f32
    float* __restrict__ out_w,              // [TOKENS, TOP_K] f32
    int* __restrict__ out_i)                // [TOKENS, TOP_K] i32
{
  __shared__ float s_scores[TOK_PER_BLOCK * NUM_EXPERTS];   // 64 KB

  const int lane = threadIdx.x;            // 0..31 (wave32)
  const int wave = threadIdx.y;            // 0..15
  const int t0   = blockIdx.x * TOK_PER_BLOCK;
  const int ts   = wave >> 2;              // token subtile 0..3 (16 tokens)
  const int n0   = (wave & 3) * 64;        // expert quad base (4 N-tiles)
  const int row  = lane & 15;
  const int hilo = lane >> 4;              // lane half

  // A fragment (16-bit A 16x32): lane holds M=row,
  // K = 16*(e/8) + 8*hilo + (e%8)  -> two 8-float chunks at +0 and +16.
  const float* Arow = hidden + (size_t)(t0 + ts * 16 + row) * HIDDEN + 8 * hilo;
  // B fragment (16-bit B 32x16): lane holds N=row, K = 16*(e/8) + 8*hilo + (e%8)
  // -> two 8-bf16 chunks at +0 and +16; tile t offset = t*16 rows.
  const uint16_t* Bbase = wbf + (size_t)(n0 + row) * HIDDEN + 8 * hilo;

  v8f acc[4] = {};
  union AF { v16bf v; uint32_t u[8]; } af;
  union BF { v16bf v; u32x4 q[2]; } bfr[4];

  for (int kk = 0; kk < HIDDEN; kk += 32) {
    __builtin_prefetch(Arow + kk + 512, 0, 0);   // A is streamed once from HBM

    f32x4 a0 = *(const f32x4*)(Arow + kk);
    f32x4 a1 = *(const f32x4*)(Arow + kk + 4);
    f32x4 a2 = *(const f32x4*)(Arow + kk + 16);
    f32x4 a3 = *(const f32x4*)(Arow + kk + 20);

#pragma unroll
    for (int t = 0; t < 4; ++t) {
      const uint16_t* Bt = Bbase + (size_t)t * 16 * HIDDEN + kk;
      bfr[t].q[0] = *(const u32x4*)(Bt);
      bfr[t].q[1] = *(const u32x4*)(Bt + 16);
    }

    af.u[0] = pack2_bf16(a0[0], a0[1]);
    af.u[1] = pack2_bf16(a0[2], a0[3]);
    af.u[2] = pack2_bf16(a1[0], a1[1]);
    af.u[3] = pack2_bf16(a1[2], a1[3]);
    af.u[4] = pack2_bf16(a2[0], a2[1]);
    af.u[5] = pack2_bf16(a2[2], a2[3]);
    af.u[6] = pack2_bf16(a3[0], a3[1]);
    af.u[7] = pack2_bf16(a3[2], a3[3]);

#pragma unroll
    for (int t = 0; t < 4; ++t)
      acc[t] = __builtin_amdgcn_wmma_f32_16x16x32_bf16(
          false, af.v, false, bfr[t].v, (short)0, acc[t], false, false);
  }

  // sigmoid + scatter to LDS.  C layout: VGPR r -> M = r + 8*hilo, N = lane&15.
#pragma unroll
  for (int t = 0; t < 4; ++t) {
#pragma unroll
    for (int r = 0; r < 8; ++r) {
      int m = ts * 16 + r + 8 * hilo;
      float s = 1.0f / (1.0f + __expf(-acc[t][r]));
      s_scores[m * NUM_EXPERTS + n0 + t * 16 + row] = s;
    }
  }
  __syncthreads();

  // ---------------- routing: wave handles 4 tokens -------------------------
  const int ebase = lane * 8;              // lane owns experts [ebase, ebase+8)

  for (int tt = 0; tt < 4; ++tt) {
    const int tl  = wave * 4 + tt;
    const int tok = t0 + tl;
    const float* srow = &s_scores[tl * NUM_EXPERTS];

    float v[8];
#pragma unroll
    for (int j = 0; j < 8; ++j) v[j] = srow[ebase + j] + bias[ebase + j];

    // per-lane top2 of 8 (m1 >= m2)
    float m1 = -1e30f, m2 = -1e30f;
#pragma unroll
    for (int j = 0; j < 8; ++j) {
      float x = v[j];
      if (x > m1) { m2 = m1; m1 = x; }
      else if (x > m2) { m2 = x; }
    }
    // merge top2 across the 4 lanes of each group (xor 1, xor 2)
#pragma unroll
    for (int off = 1; off <= 2; off <<= 1) {
      float o1 = __shfl_xor(m1, off);
      float o2 = __shfl_xor(m2, off);
      float hi = fmaxf(m1, o1);
      float lo = fminf(m1, o1);
      m1 = hi;
      m2 = fmaxf(lo, fmaxf(m2, o2));
    }
    float gs = m1 + m2;                    // group score, group = lane>>2

    // rank my group among the 8 group scores (low index wins ties)
    float gsc[N_GROUP];
#pragma unroll
    for (int j = 0; j < N_GROUP; ++j) gsc[j] = __shfl(gs, j * 4);
    int g = lane >> 2;
    int rank = 0;
#pragma unroll
    for (int j = 0; j < N_GROUP; ++j)
      rank += (gsc[j] > gsc[g]) || (gsc[j] == gsc[g] && j < g);
    bool keep = rank < TOPK_GROUP;

    // masked scores (reference multiplies by 0/1 mask)
    float mv[8];
#pragma unroll
    for (int j = 0; j < 8; ++j) mv[j] = keep ? v[j] : 0.0f;

    // iterative wave-wide top-8 (low index wins ties, matching lax.top_k)
    float pw[TOP_K];
    int pi[TOP_K];
    float wsum = 0.0f;
#pragma unroll
    for (int t = 0; t < TOP_K; ++t) {
      float bm = mv[0]; int bj = 0;
#pragma unroll
      for (int j = 1; j < 8; ++j)
        if (mv[j] > bm) { bm = mv[j]; bj = j; }
      float val = bm;
      int idx = ebase + bj;
#pragma unroll
      for (int off = 16; off > 0; off >>= 1) {
        float ov = __shfl_xor(val, off);
        int oi = __shfl_xor(idx, off);
        if (ov > val || (ov == val && oi < idx)) { val = ov; idx = oi; }
      }
      pi[t] = idx;
      float sw = srow[idx];                // sigmoid score (no bias) -> weight
      pw[t] = sw;
      wsum += sw;
      if ((idx >> 3) == lane) mv[idx & 7] = -1e30f;   // remove picked
    }

    if (lane == 0) {
      float scale = ROUTED_SCALING / (wsum + 1e-20f);
#pragma unroll
      for (int t = 0; t < TOP_K; ++t) {
        out_w[(size_t)tok * TOP_K + t] = pw[t] * scale;
        out_i[(size_t)tok * TOP_K + t] = pi[t];
      }
    }
  }
}

extern "C" void kernel_launch(void* const* d_in, const int* in_sizes, int n_in,
                              void* d_out, int out_size, void* d_ws, size_t ws_size,
                              hipStream_t stream) {
  const float* hidden = (const float*)d_in[0];
  const float* weight = (const float*)d_in[1];
  const float* bias   = (const float*)d_in[2];

  float* out_w = (float*)d_out;
  int*   out_i = (int*)d_out + (size_t)TOKENS * TOP_K;

  uint32_t* wpacked = (uint32_t*)d_ws;   // 256*7168*2 B = 3.7 MB bf16 weights
  int npairs = NUM_EXPERTS * HIDDEN / 2;
  weight_to_bf16<<<(npairs + 255) / 256, 256, 0, stream>>>(weight, wpacked, npairs);

  dim3 blk(32, 16);                      // 16 wave32 waves
  moe_gate<<<TOKENS / TOK_PER_BLOCK, blk, 0, stream>>>(
      hidden, (const uint16_t*)d_ws, bias, out_w, out_i);
}
